// Encoder_1_64733747085425
// MI455X (gfx1250) — compile-verified, matrix operand order checked by reading, not run on hardware
//
#include <hip/hip_runtime.h>

// MI455X / gfx1250, wave32, WMMA (16x16x32 f16 -> f32 accum).
//
// 2-layer biLSTM. Parallelism = 16 batch-groups x 2 dirs = 32 chains/layer,
// one wave32 per chain. Wall time ~= 2 * T * step_latency, so the design
// minimizes the per-step dependent chain: input-projection WMMAs are
// independent of h and issue first; only the small h*Whh^T WMMA, the gate
// activations (raw v_exp/v_rcp/v_tanh TRANS ops, no IEEE div expansion), and
// a same-wave LDS 16x16 transpose sit on the critical path.

typedef __attribute__((ext_vector_type(16))) _Float16 v16h;
typedef __attribute__((ext_vector_type(8)))  float    v8f;

#define B_  256
#define T_  2048
#define D_  64
#define H1_ 16
#define H2_ 8

#define LOG2E_ 1.4426950408889634f

// Raw-hardware activations: keep the recurrence critical path short.
// v_rcp_f32 instead of the IEEE div expansion (denominator >= 1 here).
__device__ __forceinline__ float sigmoidf_(float x) {
  float e = __builtin_amdgcn_exp2f(-LOG2E_ * x);   // v_exp_f32
  return __builtin_amdgcn_rcpf(1.0f + e);          // v_rcp_f32
}
__device__ __forceinline__ float tanh_(float x) {
#if __has_builtin(__builtin_amdgcn_tanhf)
  return __builtin_amdgcn_tanhf(x);                // v_tanh_f32 (CDNA5 TRANS)
#else
  float e = __builtin_amdgcn_exp2f(2.0f * LOG2E_ * x);
  return 1.0f - 2.0f * __builtin_amdgcn_rcpf(e + 1.0f);
#endif
}

__device__ __forceinline__ v8f wmma_f16(v16h a, v16h b, v8f c) {
  // (neg_a, A, neg_b, B, c_mod, C, reuse_a, reuse_b)
  return __builtin_amdgcn_wmma_f32_16x16x32_f16(false, a, false, b, (short)0, c,
                                                false, false);
}

// Build a WMMA B operand (N-tile n0..n0+15, K-chunk k0..k0+31) from row-major
// W[N][K] f32 (PyTorch weight layout: B[k][n] = W[n][k]).
// 16-bit operand layout: lane l -> col n0+(l&15), K base k0+((l>=16)?8:0);
// elems 0..7 = K base..base+7, elems 8..15 = K base+16..base+23.
__device__ __forceinline__ v16h load_B_w(const float* W, int ldw, int n0,
                                         int k0, int kvalid, int lane) {
  const float* row = W + (n0 + (lane & 15)) * ldw;
  int kb = k0 + ((lane >> 4) << 3);
  v16h r;
#pragma unroll
  for (int j = 0; j < 8; ++j) {
    int k1 = kb + j, k2 = kb + 16 + j;
    r[j]     = (k1 < kvalid) ? (_Float16)row[k1] : (_Float16)0.0f;
    r[8 + j] = (k2 < kvalid) ? (_Float16)row[k2] : (_Float16)0.0f;
  }
  return r;
}

// ---------------------------------------------------------------- layer 1 --
__global__ __launch_bounds__(32) void lstm_layer1(
    const float* __restrict__ x,
    const float* __restrict__ wih_f, const float* __restrict__ whh_f,
    const float* __restrict__ bih_f, const float* __restrict__ bhh_f,
    const float* __restrict__ wih_b, const float* __restrict__ whh_b,
    const float* __restrict__ bih_b, const float* __restrict__ bhh_b,
    _Float16* __restrict__ out1 /* [B][T][32] f16 */) {
  const int lane = threadIdx.x;
  const int grp  = blockIdx.x & 15;
  const int dir  = blockIdx.x >> 4;
  const int B0   = grp * 16;
  const int mrow = (lane >> 4) << 3;  // D-layout: upper lane-half => batch+8

  const float* wih = dir ? wih_b : wih_f;
  const float* whh = dir ? whh_b : whh_f;
  const float* bih = dir ? bih_b : bih_f;
  const float* bhh = dir ? bhh_b : bhh_f;

  // Preload weights as WMMA B operands (stay resident in VGPRs).
  v16h Bih[4][2];  // 4 gate tiles x 2 K-chunks (K=64)
  v16h Bhh[4];     // 4 gate tiles, K=16 zero-padded to 32
  float bias[4];
#pragma unroll
  for (int nt = 0; nt < 4; ++nt) {
    Bih[nt][0] = load_B_w(wih, D_, nt * 16, 0,  D_,  lane);
    Bih[nt][1] = load_B_w(wih, D_, nt * 16, 32, D_,  lane);
    Bhh[nt]    = load_B_w(whh, H1_, nt * 16, 0, H1_, lane);
    int n = nt * 16 + (lane & 15);
    bias[nt] = bih[n] + bhh[n];
  }

  __shared__ float sh[16 * 16];  // h transpose buffer (single wave)

  float c[8], h[8];
  v16h ah;  // h in WMMA A layout (K=16 padded to 32)
#pragma unroll
  for (int r = 0; r < 8; ++r) { c[r] = 0.0f; h[r] = 0.0f; }
#pragma unroll
  for (int j = 0; j < 16; ++j) ah[j] = (_Float16)0.0f;

  const int bA = B0 + (lane & 15);  // A-operand batch row for this lane
  const float* xrow_base = x + (size_t)bA * T_ * D_;

  for (int t = 0; t < T_; ++t) {
    const int tx = dir ? (T_ - 1 - t) : t;
    const float* xr = xrow_base + (size_t)tx * D_;

    // Prefetch x 8 steps ahead (global_prefetch_b8) to hide HBM latency.
    int tp = t + 8;
    if (tp < T_) {
      int txp = dir ? (T_ - 1 - tp) : tp;
      const float* xp = xrow_base + (size_t)txp * D_ + ((lane >> 4) << 5);
      __builtin_prefetch(xp, 0, 1);
    }

    // A operands for x[16 batch, 64] (f32 -> f16 in-register).
    v16h a0, a1;
    {
      int kb = (lane >> 4) << 3;
#pragma unroll
      for (int j = 0; j < 8; ++j) {
        a0[j]     = (_Float16)xr[kb + j];
        a0[8 + j] = (_Float16)xr[kb + 16 + j];
        a1[j]     = (_Float16)xr[32 + kb + j];
        a1[8 + j] = (_Float16)xr[32 + kb + 16 + j];
      }
    }

    // Gate pre-activations: tile 0..3 = i,f,g,o (each 16 wide).
    v8f acc[4];
#pragma unroll
    for (int nt = 0; nt < 4; ++nt) {
      v8f a;
#pragma unroll
      for (int j = 0; j < 8; ++j) a[j] = bias[nt];
      a = wmma_f16(a0, Bih[nt][0], a);  // independent of h-chain
      a = wmma_f16(a1, Bih[nt][1], a);
      a = wmma_f16(ah, Bhh[nt], a);     // recurrent (critical path)
      acc[nt] = a;
    }

    // Elementwise cell update (lane = hidden unit, vgpr = batch).
#pragma unroll
    for (int r = 0; r < 8; ++r) {
      float ig = sigmoidf_(acc[0][r]);
      float fg = sigmoidf_(acc[1][r]);
      float gg = tanh_(acc[2][r]);
      float og = sigmoidf_(acc[3][r]);
      c[r] = fg * c[r] + ig * gg;
      h[r] = og * tanh_(c[r]);
    }

    // Emit h as f16 into out1[B][T][32] at [:, tx, dir*16 + n].
#pragma unroll
    for (int r = 0; r < 8; ++r) {
      size_t o = ((size_t)(B0 + r + mrow) * T_ + tx) * 32 + dir * 16 + (lane & 15);
      out1[o] = (_Float16)h[r];
    }

    // Transpose h (D layout) -> A layout via LDS for the next step.
    __syncthreads();
#pragma unroll
    for (int r = 0; r < 8; ++r) sh[(r + mrow) * 16 + (lane & 15)] = h[r];
    __syncthreads();
    {
      int m  = lane & 15;
      int kb = (lane >> 4) << 3;
#pragma unroll
      for (int j = 0; j < 8; ++j) {
        ah[j]     = (_Float16)sh[m * 16 + kb + j];
        ah[8 + j] = (_Float16)0.0f;  // K = 16..31 padding
      }
    }
  }
}

// ---------------------------------------------------------------- layer 2 --
__global__ __launch_bounds__(32) void lstm_layer2(
    const _Float16* __restrict__ in1 /* [B][T][32] f16 */,
    const float* __restrict__ wih_f, const float* __restrict__ whh_f,
    const float* __restrict__ bih_f, const float* __restrict__ bhh_f,
    const float* __restrict__ wih_b, const float* __restrict__ whh_b,
    const float* __restrict__ bih_b, const float* __restrict__ bhh_b,
    float* __restrict__ out /* [B][T][16] f32 */) {
  const int lane = threadIdx.x;
  const int grp  = blockIdx.x & 15;
  const int dir  = blockIdx.x >> 4;
  const int B0   = grp * 16;
  const int mrow = (lane >> 4) << 3;

  const float* wih = dir ? wih_b : wih_f;
  const float* whh = dir ? whh_b : whh_f;
  const float* bih = dir ? bih_b : bih_f;
  const float* bhh = dir ? bhh_b : bhh_f;

  // 32 gates = 2 N-tiles: tile0 = [i(8)|f(8)], tile1 = [g(8)|o(8)].
  v16h Bih[2], Bhh[2];
  float bias[2];
#pragma unroll
  for (int nt = 0; nt < 2; ++nt) {
    Bih[nt] = load_B_w(wih, 2 * H1_, nt * 16, 0, 2 * H1_, lane);  // K=32
    Bhh[nt] = load_B_w(whh, H2_,     nt * 16, 0, H2_,     lane);  // K=8 padded
    int n = nt * 16 + (lane & 15);
    bias[nt] = bih[n] + bhh[n];
  }

  __shared__ float sh[16 * 8];

  float c[8], h[8];
  v16h ah;
#pragma unroll
  for (int r = 0; r < 8; ++r) { c[r] = 0.0f; h[r] = 0.0f; }
#pragma unroll
  for (int j = 0; j < 16; ++j) ah[j] = (_Float16)0.0f;

  const _Float16* inrow_base = in1 + (size_t)(B0 + (lane & 15)) * T_ * 32;

  for (int t = 0; t < T_; ++t) {
    const int tx = dir ? (T_ - 1 - t) : t;
    const _Float16* ir = inrow_base + (size_t)tx * 32;

    int tp = t + 8;
    if (tp < T_) {
      int txp = dir ? (T_ - 1 - tp) : tp;
      __builtin_prefetch(inrow_base + (size_t)txp * 32, 0, 1);
    }

    // A operand: in1 tile [16 batch, K=32] already f16.
    v16h a0;
    {
      int kb = (lane >> 4) << 3;
#pragma unroll
      for (int j = 0; j < 8; ++j) {
        a0[j]     = ir[kb + j];
        a0[8 + j] = ir[kb + 16 + j];
      }
    }

    v8f acc[2];
#pragma unroll
    for (int nt = 0; nt < 2; ++nt) {
      v8f a;
#pragma unroll
      for (int j = 0; j < 8; ++j) a[j] = bias[nt];
      a = wmma_f16(a0, Bih[nt], a);
      a = wmma_f16(ah, Bhh[nt], a);
      acc[nt] = a;
    }

    // Elementwise: lane n<8 holds i,g; n>=8 holds f,o -> shuffle f,o down.
    const bool low = (lane & 15) < 8;
#pragma unroll
    for (int r = 0; r < 8; ++r) {
      float s0 = sigmoidf_(acc[0][r]);                           // i | f
      float s1 = low ? tanh_(acc[1][r]) : sigmoidf_(acc[1][r]);  // g | o
      float fg = __shfl_xor(s0, 8);  // lanes n<8 receive f
      float og = __shfl_xor(s1, 8);  // lanes n<8 receive o
      c[r] = fg * c[r] + s0 * s1;    // f*c + i*g   (valid on low lanes)
      h[r] = og * tanh_(c[r]);
    }

    // Final output f32 [B][T][16]: fwd -> [0:8], bwd -> [8:16].
    if (low) {
#pragma unroll
      for (int r = 0; r < 8; ++r) {
        size_t o = ((size_t)(B0 + r + mrow) * T_ + tx) * 16 + dir * 8 + (lane & 7);
        out[o] = h[r];
      }
    }

    // Transpose h [16 batch, 8 hid] -> A layout (K=8, rest zero).
    __syncthreads();
    if (low) {
#pragma unroll
      for (int r = 0; r < 8; ++r) sh[(r + mrow) * 8 + (lane & 7)] = h[r];
    }
    __syncthreads();
    {
      int m = lane & 15;
      bool lo16 = lane < 16;
#pragma unroll
      for (int j = 0; j < 8; ++j) {
        ah[j]     = lo16 ? (_Float16)sh[m * 8 + j] : (_Float16)0.0f;
        ah[8 + j] = (_Float16)0.0f;
      }
    }
  }
}

// ------------------------------------------------------------------ launch --
extern "C" void kernel_launch(void* const* d_in, const int* in_sizes, int n_in,
                              void* d_out, int out_size, void* d_ws, size_t ws_size,
                              hipStream_t stream) {
  const float* x = (const float*)d_in[0];
  _Float16* out1 = (_Float16*)d_ws;  // [B][T][32] f16 = 32 MiB

  lstm_layer1<<<32, 32, 0, stream>>>(
      x,
      (const float*)d_in[1], (const float*)d_in[2],
      (const float*)d_in[3], (const float*)d_in[4],
      (const float*)d_in[5], (const float*)d_in[6],
      (const float*)d_in[7], (const float*)d_in[8],
      out1);

  lstm_layer2<<<32, 32, 0, stream>>>(
      out1,
      (const float*)d_in[9],  (const float*)d_in[10],
      (const float*)d_in[11], (const float*)d_in[12],
      (const float*)d_in[13], (const float*)d_in[14],
      (const float*)d_in[15], (const float*)d_in[16],
      (float*)d_out);
}
